// CrossModalAttention_48558900248898
// MI455X (gfx1250) — matrix operand order repaired and checked by reading, not compiled, not run on hardware
//
#include <hip/hip_runtime.h>
#include <hip/hip_bf16.h>

typedef __attribute__((ext_vector_type(16))) __bf16 v16bf;
typedef __attribute__((ext_vector_type(8)))  __bf16 bf16x8;
typedef __attribute__((ext_vector_type(2)))  __bf16 bf16x2;
typedef __attribute__((ext_vector_type(8)))  float  v8f;

#define CC    256
#define HH    128
#define WWID  128
#define HWSZ  (HH * WWID)
#define BB    2
#define M_TILE 64
#define N_TILE 128
#define K_STEP 32

// ---------------------------------------------------------------------------
// Kernel 1: fused Q/K/V 1x1-conv projections as bf16 WMMA GEMMs.
// out[proj][b][m][n] = sum_k w[m][k] * x[b][k][n] + bias[m]
//
// LDS tiles live in WMMA-fragment order ([subtile][lane][element]):
//   - lane fragment fetch = 2x ds_load_b128
//   - A staging: 8 consecutive k -> one ds_store_b128
//   - B staging: (k,k+1) bf16 pairs -> ds_store_b32 (merged to 2addr by LLVM)
// 64x128 block tile, 8 waves, each wave = 1 A-frag x 4 B-frags = 4 WMMA/step.
// Double-buffered LDS -> one barrier per K-step; loads overlap WMMAs.
// ---------------------------------------------------------------------------
__global__ __launch_bounds__(256) void qkv_proj_wmma(
    const float* __restrict__ blue, const float* __restrict__ white,
    const float* __restrict__ qw, const float* __restrict__ qb,
    const float* __restrict__ kw, const float* __restrict__ kb,
    const float* __restrict__ vw, const float* __restrict__ vb,
    float* __restrict__ qkv)
{
  const int proj = blockIdx.z % 3;
  const int b    = blockIdx.z / 3;
  const float* x    = (proj == 0 ? blue : white) + (size_t)b * CC * HWSZ;
  const float* w    = (proj == 0 ? qw : (proj == 1 ? kw : vw));
  const float* bias = (proj == 0 ? qb : (proj == 1 ? kb : vb));
  float* out = qkv + ((size_t)proj * BB + b) * CC * HWSZ;

  const int n0 = blockIdx.x * N_TILE;   // pixel tile
  const int m0 = blockIdx.y * M_TILE;   // out-channel tile

  // Fragment-ordered, double-buffered LDS.
  __shared__ __attribute__((aligned(64))) __bf16 aLds[2][4 * 32 * 16];  // W 64x32
  __shared__ __attribute__((aligned(64))) __bf16 bLds[2][8 * 32 * 16];  // X 32x128

  const int tid  = threadIdx.x;
  const int wave = tid >> 5;
  const int lane = tid & 31;
  const int half = lane >> 4;
  const int lrow = lane & 15;

  const int wave_m = wave & 3;    // 4 M subtiles of 16
  const int wave_n = wave >> 2;   // 2 N groups of 64 (4 subtiles each)

  // ---- Per-thread staging geometry (loop-invariant) ----
  // A: thread -> (row mm, 8-aligned k group kg): 64 x 32 = 2048 elems
  const int a_mm = tid >> 2;
  const int a_kg = (tid & 3) << 3;
  const float* aSrc = &w[(size_t)(m0 + a_mm) * CC + a_kg];
  const int a_half = a_kg >> 4;
  const int a_off  = (((a_mm >> 4) * 32) + (a_mm & 15) + (a_half << 4)) * 16 +
                     (a_kg & 15);
  // B: thread -> (even k pair kk, 8-wide n group nnb): 32 x 128 = 4096 elems
  const int b_kk  = (tid >> 4) << 1;
  const int b_nnb = (tid & 15) << 3;
  const float* bSrc = &x[(size_t)b_kk * HWSZ + (n0 + b_nnb)];
  const int b_e = b_kk & 15;
  int b_off[8];
  #pragma unroll
  for (int q = 0; q < 8; q++) {
    int nn = b_nnb + q;
    b_off[q] = (((nn >> 4) * 32) + (nn & 15) + ((b_kk >> 4) << 4)) * 16 + b_e;
  }

  // ---- Staging helper ----
  auto stage = [&](int step, int bufi) {
    // A tile: 2x float4 -> 8 packed bf16 -> one ds_store_b128
    const float* ap = aSrc + step * K_STEP;
    float4 ar0 = *(const float4*)(ap);
    float4 ar1 = *(const float4*)(ap + 4);
    bf16x8 av;
    av[0] = (__bf16)ar0.x; av[1] = (__bf16)ar0.y;
    av[2] = (__bf16)ar0.z; av[3] = (__bf16)ar0.w;
    av[4] = (__bf16)ar1.x; av[5] = (__bf16)ar1.y;
    av[6] = (__bf16)ar1.z; av[7] = (__bf16)ar1.w;
    *(bf16x8*)&aLds[bufi][a_off] = av;

    // B tile: rows k and k+1, 8 n each -> 8 bf16x2 pair stores
    const float* bp = bSrc + (size_t)step * K_STEP * HWSZ;
    float4 br0 = *(const float4*)(bp);
    float4 br1 = *(const float4*)(bp + 4);
    float4 br2 = *(const float4*)(bp + HWSZ);
    float4 br3 = *(const float4*)(bp + HWSZ + 4);
    float r0[8] = {br0.x, br0.y, br0.z, br0.w, br1.x, br1.y, br1.z, br1.w};
    float r1[8] = {br2.x, br2.y, br2.z, br2.w, br3.x, br3.y, br3.z, br3.w};
    #pragma unroll
    for (int q = 0; q < 8; q++) {
      bf16x2 p;
      p[0] = (__bf16)r0[q];
      p[1] = (__bf16)r1[q];
      *(bf16x2*)&bLds[bufi][b_off[q]] = p;
    }
  };

  const int aFragOff = (wave_m * 32 + lane) * 16;
  int bFragOff[4];
  #pragma unroll
  for (int q = 0; q < 4; q++)
    bFragOff[q] = ((wave_n * 4 + q) * 32 + lane) * 16;

  v8f acc[4] = {};

  stage(0, 0);

  #pragma unroll
  for (int s = 0; s < CC / K_STEP; s++) {
    const int buf = s & 1;
    __syncthreads();
    if (s + 1 < CC / K_STEP) stage(s + 1, buf ^ 1);

    v16bf afrag = *(const v16bf*)&aLds[buf][aFragOff];
    v16bf bf0   = *(const v16bf*)&bLds[buf][bFragOff[0]];
    v16bf bf1   = *(const v16bf*)&bLds[buf][bFragOff[1]];
    v16bf bf2   = *(const v16bf*)&bLds[buf][bFragOff[2]];
    v16bf bf3   = *(const v16bf*)&bLds[buf][bFragOff[3]];

    acc[0] = __builtin_amdgcn_wmma_f32_16x16x32_bf16(false, afrag, false, bf0,
                                                     (short)0, acc[0], false, false);
    acc[1] = __builtin_amdgcn_wmma_f32_16x16x32_bf16(false, afrag, false, bf1,
                                                     (short)0, acc[1], false, false);
    acc[2] = __builtin_amdgcn_wmma_f32_16x16x32_bf16(false, afrag, false, bf2,
                                                     (short)0, acc[2], false, false);
    acc[3] = __builtin_amdgcn_wmma_f32_16x16x32_bf16(false, afrag, false, bf3,
                                                     (short)0, acc[3], false, false);
  }

  // Epilogue: C/D layout -> VGPR r holds M = r + 8*half, N = lane%16.
  const int m_base = m0 + wave_m * 16 + 8 * half;
  const int n_base = n0 + wave_n * 64 + lrow;
  #pragma unroll
  for (int r = 0; r < 8; r++) {
    int m = m_base + r;
    float bv = bias[m];
    float* op = out + (size_t)m * HWSZ + n_base;
    op[0]  = acc[0][r] + bv;
    op[16] = acc[1][r] + bv;
    op[32] = acc[2][r] + bv;
    op[48] = acc[3][r] + bv;
  }
}

// ---------------------------------------------------------------------------
// Kernel 2: per-channel neighborhood token attention.
// One block per (b, i, j) token; thread = channel. K/V re-reads hit L2.
// ---------------------------------------------------------------------------
__global__ __launch_bounds__(256) void token_attn(
    const float* __restrict__ qkv, float* __restrict__ out)
{
  const int nh = HH / 4, nw = WWID / 4;     // 32 x 32 token grid
  int blk = blockIdx.x;
  const int j = blk % nw; blk /= nw;
  const int i = blk % nh; blk /= nh;
  const int b = blk;
  const int c = threadIdx.x;

  const float* Q = qkv + ((size_t)0 * BB + b) * CC * HWSZ + (size_t)c * HWSZ;
  const float* K = qkv + ((size_t)1 * BB + b) * CC * HWSZ + (size_t)c * HWSZ;
  const float* V = qkv + ((size_t)2 * BB + b) * CC * HWSZ + (size_t)c * HWSZ;

  // Query token: 4 rows of float4.
  float4 q[4];
  #pragma unroll
  for (int ti = 0; ti < 4; ti++)
    q[ti] = *(const float4*)(Q + (size_t)(i * 4 + ti) * WWID + j * 4);

  // Scores vs 9 neighbor tokens (masked), SCALE = (256/16)^-0.5 = 0.25.
  float s[9];
  #pragma unroll
  for (int n = 0; n < 9; n++) {
    const int ii = i + n / 3 - 1;
    const int jj = j + n % 3 - 1;
    if (ii >= 0 && ii < nh && jj >= 0 && jj < nw) {
      float a = 0.0f;
      #pragma unroll
      for (int ti = 0; ti < 4; ti++) {
        float4 kt = *(const float4*)(K + (size_t)(ii * 4 + ti) * WWID + jj * 4);
        a += q[ti].x * kt.x + q[ti].y * kt.y + q[ti].z * kt.z + q[ti].w * kt.w;
      }
      s[n] = a * 0.25f;
    } else {
      s[n] = -INFINITY;
    }
  }

  // Softmax over the 9 neighbors.
  float m = s[0];
  #pragma unroll
  for (int n = 1; n < 9; n++) m = fmaxf(m, s[n]);
  float sum = 0.0f;
  #pragma unroll
  for (int n = 0; n < 9; n++) { s[n] = __expf(s[n] - m); sum += s[n]; }
  const float inv = 1.0f / sum;

  // Weighted V accumulation.
  float4 o[4] = {};
  #pragma unroll
  for (int n = 0; n < 9; n++) {
    const int ii = i + n / 3 - 1;
    const int jj = j + n % 3 - 1;
    if (ii >= 0 && ii < nh && jj >= 0 && jj < nw) {
      const float wn = s[n] * inv;
      #pragma unroll
      for (int ti = 0; ti < 4; ti++) {
        float4 vt = *(const float4*)(V + (size_t)(ii * 4 + ti) * WWID + jj * 4);
        o[ti].x += wn * vt.x; o[ti].y += wn * vt.y;
        o[ti].z += wn * vt.z; o[ti].w += wn * vt.w;
      }
    }
  }

  // Untokenized store: out[b, c, h, w].
  float* O = out + ((size_t)b * CC + c) * HWSZ;
  #pragma unroll
  for (int ti = 0; ti < 4; ti++)
    *(float4*)(O + (size_t)(i * 4 + ti) * WWID + j * 4) = o[ti];
}

// ---------------------------------------------------------------------------
extern "C" void kernel_launch(void* const* d_in, const int* in_sizes, int n_in,
                              void* d_out, int out_size, void* d_ws, size_t ws_size,
                              hipStream_t stream) {
  const float* blue  = (const float*)d_in[0];
  const float* white = (const float*)d_in[1];
  const float* qw    = (const float*)d_in[2];
  const float* qb    = (const float*)d_in[3];
  const float* kw    = (const float*)d_in[4];
  const float* kb    = (const float*)d_in[5];
  const float* vw    = (const float*)d_in[6];
  const float* vb    = (const float*)d_in[7];

  float* qkv = (float*)d_ws;   // [3][B][C][H*W] fp32 = 96 MiB scratch

  dim3 gProj(HWSZ / N_TILE, CC / M_TILE, 3 * BB);   // 128 x 4 x 6
  qkv_proj_wmma<<<gProj, 256, 0, stream>>>(blue, white, qw, qb, kw, kb, vw, vb, qkv);

  const int nTok = BB * (HH / 4) * (WWID / 4);      // 2048 token blocks
  token_attn<<<nTok, 256, 0, stream>>>(qkv, (float*)d_out);
}
